// Transformer_84104049590755
// MI455X (gfx1250) — matrix operand order repaired and checked by reading, not compiled, not run on hardware
//
#include <hip/hip_runtime.h>
#include <hip/hip_bf16.h>
#include <math.h>

// ---------------------------------------------------------------------------
// GPT-2-small-ish forward pass for MI455X (gfx1250, wave32, WMMA).
// bf16 weights (pre-converted once), fp32 activations converted in-flight,
// v_wmma_f32_16x16x32_bf16 everywhere, flash attention, fused epilogues.
// B tiles staged into LDS with GLOBAL_LOAD_ASYNC_TO_LDS_B128 (ASYNCcnt) when
// the toolchain exposes the builtin; next-tile global_prefetch_b8 issued.
// Workload is HBM-bound (412 MB logits + 120 MB bf16 weights, 23.3 TB/s).
// ---------------------------------------------------------------------------

typedef __attribute__((ext_vector_type(16))) __bf16 v16bf;
typedef __attribute__((ext_vector_type(8)))  float  v8f;
typedef int v4i_t __attribute__((vector_size(16)));  // matches builtin's V4i

#define SEQ  2048
#define VOC  50257
#define VOCP 50304            // VOC padded to a multiple of 128 (and of 8)
#define DIM  768
#define NH   12
#define NL   4
#define HS   64
#define FF   3072
#define QKVSTR (NH * 3 * HS)  // 2304: row stride of qkv buffer [S, H, 192]

#if defined(__gfx1250__) && __has_builtin(__builtin_amdgcn_global_load_async_to_lds_b128)
#define HAVE_ASYNC_LDS 1
#else
#define HAVE_ASYNC_LDS 0
#endif

__device__ __forceinline__ void async_wait_zero() {
#if defined(__gfx1250__)
#if __has_builtin(__builtin_amdgcn_s_wait_asynccnt)
  __builtin_amdgcn_s_wait_asynccnt(0);
#else
  asm volatile("s_wait_asynccnt 0" ::: "memory");
#endif
#endif
}

// --------------------------- weight conversion -----------------------------
__global__ __launch_bounds__(256) void f32_to_bf16(const float* __restrict__ in,
                                                   __bf16* __restrict__ out, int n) {
  for (int i = blockIdx.x * 256 + threadIdx.x; i < n; i += gridDim.x * 256)
    out[i] = (__bf16)in[i];
}

// convert with row padding: out[r, 0..npad) = in[r, 0..n) zero-extended
__global__ __launch_bounds__(256) void f32_to_bf16_pad(const float* __restrict__ in,
                                                       __bf16* __restrict__ out,
                                                       int rows, int n, int npad) {
  long total = (long)rows * npad;
  for (long i = blockIdx.x * 256L + threadIdx.x; i < total; i += gridDim.x * 256L) {
    int r = (int)(i / npad), c = (int)(i % npad);
    out[i] = (c < n) ? (__bf16)in[(size_t)r * n + c] : (__bf16)0.f;
  }
}

// ------------------------------- embedding ---------------------------------
__global__ __launch_bounds__(256) void embed_kernel(const int* __restrict__ tokens,
                                                    const float* __restrict__ embed_w,
                                                    const float* __restrict__ posn_w,
                                                    float* __restrict__ x) {
  int s = blockIdx.x;
  int tok = tokens[s];
  for (int i = threadIdx.x; i < DIM; i += 256)
    x[(size_t)s * DIM + i] = embed_w[(size_t)tok * DIM + i] + posn_w[(size_t)s * DIM + i];
}

// ------------------------------ WMMA GEMM ----------------------------------
// C[M,N] = act(A[M,K] (f32) @ B[K,N] (bf16, leading dim ldb) + bias)
// Block: 256 thr = 8 waves. Block tile 64(M) x 128(N); wave (wm,wn) owns a
// 32x32 patch = 4 accumulators (2 A-frags x 2 B-frags reused).
// OP: 0 = none, 1 = +bias, 2 = +bias then exact GELU.
template <int OP>
__global__ __launch_bounds__(256) void wmma_gemm(const float* __restrict__ A,
                                                 const __bf16* __restrict__ B,
                                                 const float* __restrict__ bias,
                                                 float* __restrict__ C,
                                                 int M, int N, int K, int ldb) {
  __shared__ __bf16 sA[64][32];
  __shared__ __bf16 sB[32][128];

  const int tid  = threadIdx.x;
  const int lane = tid & 31;
  const int wave = tid >> 5;
  const int wm   = (wave >> 2) * 32;  // 0 or 32
  const int wn   = (wave & 3) * 32;   // 0,32,64,96
  const int row0 = blockIdx.y * 64;
  const int col0 = blockIdx.x * 128;
  // async path needs: full 128-col tile inside padded ld, 16B-aligned rows
  const bool async_ok = (col0 + 128 <= ldb) && ((ldb & 7) == 0);

  const v8f vz = {};
  v8f acc[2][2] = {{vz, vz}, {vz, vz}};

  for (int k0 = 0; k0 < K; k0 += 32) {
    // ---- stage A tile (64x32, f32 -> bf16): 2048 elems, 8 per thread ----
    for (int i = tid; i < 64 * 32; i += 256) {
      int m = i >> 5, kk = i & 31;
      sA[m][kk] = (__bf16)A[(size_t)(row0 + m) * K + (k0 + kk)];
    }
    // ---- stage B tile (32x128 bf16 = 8KB) ----
#if HAVE_ASYNC_LDS
    if (async_ok) {
      // 512 x 16B chunks, 2 per thread, ASYNCcnt-tracked direct-to-LDS.
#pragma unroll
      for (int c = 0; c < 2; ++c) {
        int idx = tid + c * 256;
        int kk  = idx >> 4;          // 0..31
        int cb  = (idx & 15) << 3;   // column offset in elems (8 bf16 = 16B)
        const __bf16* g = B + (size_t)(k0 + kk) * ldb + col0 + cb;
        __builtin_amdgcn_global_load_async_to_lds_b128(
            (__attribute__((address_space(1))) v4i_t*)(uintptr_t)g,
            (__attribute__((address_space(3))) v4i_t*)(uintptr_t)&sB[kk][cb],
            0, 0);
      }
      async_wait_zero();
    } else
#endif
    {
      for (int i = tid; i < 32 * 128; i += 256) {
        int kk = i >> 7, n = i & 127;
        int gn = col0 + n;
        sB[kk][n] = (gn < ldb) ? B[(size_t)(k0 + kk) * ldb + gn] : (__bf16)0.f;
      }
    }
    // prefetch next K-step of the B stream into L2/WGP$
    if (k0 + 32 < K) {
      int kk = tid >> 3, cb = (tid & 7) << 4;
      __builtin_prefetch(B + (size_t)(k0 + 32 + kk) * ldb + col0 + cb, 0, 1);
    }
    __syncthreads();

    // ---- fragments (contiguous k = 16*(lane>>4) + i, same order in A and B) ----
    v16bf afr[2], bfr[2];
#pragma unroll
    for (int mi = 0; mi < 2; ++mi) {
      int m  = wm + mi * 16 + (lane & 15);
      int kh = (lane >> 4) * 16;
#pragma unroll
      for (int i = 0; i < 8; ++i) afr[mi][i]     = sA[m][kh + i];
#pragma unroll
      for (int i = 0; i < 8; ++i) afr[mi][8 + i] = sA[m][kh + 8 + i];
    }
#pragma unroll
    for (int ni = 0; ni < 2; ++ni) {
      int n  = wn + ni * 16 + (lane & 15);
      int kb = (lane >> 4) * 16;
#pragma unroll
      for (int i = 0; i < 16; ++i) bfr[ni][i] = sB[kb + i][n];
    }
#pragma unroll
    for (int mi = 0; mi < 2; ++mi)
#pragma unroll
      for (int ni = 0; ni < 2; ++ni)
        acc[mi][ni] = __builtin_amdgcn_wmma_f32_16x16x32_bf16(
            false, afr[mi], false, bfr[ni], (short)0, acc[mi][ni], false, false);
    __syncthreads();
  }

  // ---- epilogue: C/D layout VGPR r -> M = r + 8*(lane>=16); N = lane&15 ----
#pragma unroll
  for (int ni = 0; ni < 2; ++ni) {
    int n = col0 + wn + ni * 16 + (lane & 15);
    if (n < N) {
      float bv = (OP >= 1) ? bias[n] : 0.f;
#pragma unroll
      for (int mi = 0; mi < 2; ++mi) {
#pragma unroll
        for (int r = 0; r < 8; ++r) {
          int m = row0 + wm + mi * 16 + r + ((lane >> 4) << 3);
          float v = acc[mi][ni][r];
          if (OP >= 1) v += bv;
          if (OP == 2) v = 0.5f * v * (1.f + erff(v * 0.70710678118654752f));
          C[(size_t)m * N + n] = v;
        }
      }
    }
  }
}

// --------------------------- flash attention -------------------------------
// One wave per (head, 16-query block). qkv: [S, H, 192] f32 (q|k|v per head).
__global__ __launch_bounds__(32) void attn_kernel(const float* __restrict__ qkv,
                                                  float* __restrict__ aout) {
  __shared__ float  s_sc[16][32];
  __shared__ __bf16 s_p[16][32];
  __shared__ float  s_fac[16];
  __shared__ float  s_l[16];

  const int h    = blockIdx.x;
  const int q0   = blockIdx.y * 16;
  const int lane = threadIdx.x;
  const float scale = 0.125f;  // 1/sqrt(64)

  v16bf aq[2];
  {
    int m  = lane & 15;
    int kh = (lane >> 4) * 16;
    const float* qrow = qkv + (size_t)(q0 + m) * QKVSTR + h * 3 * HS;
#pragma unroll
    for (int c = 0; c < 2; ++c) {
#pragma unroll
      for (int i = 0; i < 8; ++i) aq[c][i]     = (__bf16)qrow[c * 32 + kh + i];
#pragma unroll
      for (int i = 0; i < 8; ++i) aq[c][8 + i] = (__bf16)qrow[c * 32 + kh + 8 + i];
    }
  }

  const v8f vz = {};
  v8f o[4] = {vz, vz, vz, vz};
  float m_i = -INFINITY, l_i = 0.f;

  const int ntiles = (q0 + 16 + 31) >> 5;
  for (int t = 0; t < ntiles; ++t) {
    const int jb = t * 32;

    v8f sc[2] = {vz, vz};
#pragma unroll
    for (int sub = 0; sub < 2; ++sub) {
      int j = jb + sub * 16 + (lane & 15);
      if (j > SEQ - 1) j = SEQ - 1;
      const float* krow = qkv + (size_t)j * QKVSTR + h * 3 * HS + HS;
      int kb = (lane >> 4) * 16;
      v16bf bk;
#pragma unroll
      for (int i = 0; i < 16; ++i) bk[i] = (__bf16)krow[kb + i];
      sc[sub] = __builtin_amdgcn_wmma_f32_16x16x32_bf16(false, aq[0], false, bk,
                                                        (short)0, sc[sub], false, false);
#pragma unroll
      for (int i = 0; i < 16; ++i) bk[i] = (__bf16)krow[32 + kb + i];
      sc[sub] = __builtin_amdgcn_wmma_f32_16x16x32_bf16(false, aq[1], false, bk,
                                                        (short)0, sc[sub], false, false);
    }

#pragma unroll
    for (int r = 0; r < 8; ++r) {
      int m = r + ((lane >> 4) << 3);
      s_sc[m][lane & 15]        = sc[0][r];
      s_sc[m][16 + (lane & 15)] = sc[1][r];
    }
    __syncthreads();

    if (lane < 16) {
      int iglob = q0 + lane;
      float vals[32];
      float tmax = -INFINITY;
#pragma unroll
      for (int c = 0; c < 32; ++c) {
        int j = jb + c;
        float v = (j <= iglob) ? s_sc[lane][c] * scale : -INFINITY;
        vals[c] = v;
        tmax = fmaxf(tmax, v);
      }
      float new_m = fmaxf(m_i, tmax);
      float fac   = __expf(m_i - new_m);
      float psum  = 0.f;
#pragma unroll
      for (int c = 0; c < 32; ++c) {
        float p = __expf(vals[c] - new_m);
        psum += p;
        s_p[lane][c] = (__bf16)p;
      }
      l_i = l_i * fac + psum;
      m_i = new_m;
      s_fac[lane] = fac;
    }
    __syncthreads();

#pragma unroll
    for (int r = 0; r < 8; ++r) {
      int m = r + ((lane >> 4) << 3);
      float f = s_fac[m];
#pragma unroll
      for (int nf = 0; nf < 4; ++nf) o[nf][r] *= f;
    }

    v16bf ap;
    {
      int m  = lane & 15;
      int kh = (lane >> 4) * 16;
#pragma unroll
      for (int i = 0; i < 8; ++i) ap[i]     = s_p[m][kh + i];
#pragma unroll
      for (int i = 0; i < 8; ++i) ap[8 + i] = s_p[m][kh + 8 + i];
    }
#pragma unroll
    for (int nf = 0; nf < 4; ++nf) {
      int ncol = nf * 16 + (lane & 15);
      int kb   = (lane >> 4) * 16;
      v16bf bv;
#pragma unroll
      for (int i = 0; i < 16; ++i) {
        int j = jb + kb + i;
        if (j > SEQ - 1) j = SEQ - 1;  // p==0 there
        bv[i] = (__bf16)qkv[(size_t)j * QKVSTR + h * 3 * HS + 2 * HS + ncol];
      }
      o[nf] = __builtin_amdgcn_wmma_f32_16x16x32_bf16(false, ap, false, bv,
                                                      (short)0, o[nf], false, false);
    }
    __syncthreads();
  }

  if (lane < 16) s_l[lane] = l_i;
  __syncthreads();

#pragma unroll
  for (int r = 0; r < 8; ++r) {
    int m = r + ((lane >> 4) << 3);
    float inv = 1.f / s_l[m];
    int gq = q0 + m;
#pragma unroll
    for (int nf = 0; nf < 4; ++nf) {
      int n = nf * 16 + (lane & 15);
      aout[(size_t)gq * DIM + h * HS + n] = o[nf][r] * inv;
    }
  }
}

// --------------------- fused LayerNorm + residual add ----------------------
__global__ __launch_bounds__(256) void ln_residual(const float* __restrict__ t,
                                                   const float* __restrict__ w,
                                                   const float* __restrict__ b,
                                                   float* __restrict__ x) {
  __shared__ float sred[256];
  const int row = blockIdx.x;
  const int tid = threadIdx.x;
  const float* tr = t + (size_t)row * DIM;
  float* xr = x + (size_t)row * DIM;

  float s = 0.f;
  for (int i = tid; i < DIM; i += 256) s += tr[i];
  sred[tid] = s; __syncthreads();
  for (int k = 128; k > 0; k >>= 1) { if (tid < k) sred[tid] += sred[tid + k]; __syncthreads(); }
  float mu = sred[0] * (1.f / DIM);
  __syncthreads();

  float v = 0.f;
  for (int i = tid; i < DIM; i += 256) { float d = tr[i] - mu; v += d * d; }
  sred[tid] = v; __syncthreads();
  for (int k = 128; k > 0; k >>= 1) { if (tid < k) sred[tid] += sred[tid + k]; __syncthreads(); }
  float rstd = rsqrtf(sred[0] * (1.f / DIM) + 1e-5f);

  for (int i = tid; i < DIM; i += 256)
    xr[i] += (tr[i] - mu) * rstd * w[i] + b[i];
}

// --------------------------- row softmax (in place) ------------------------
__global__ __launch_bounds__(256) void softmax_rows(float* __restrict__ out, int N) {
  __shared__ float sred[256];
  float* row = out + (size_t)blockIdx.x * N;
  const int tid = threadIdx.x;

  float mx = -INFINITY;
  for (int i = tid; i < N; i += 256) mx = fmaxf(mx, row[i]);
  sred[tid] = mx; __syncthreads();
  for (int k = 128; k > 0; k >>= 1) { if (tid < k) sred[tid] = fmaxf(sred[tid], sred[tid + k]); __syncthreads(); }
  mx = sred[0]; __syncthreads();

  float sum = 0.f;
  for (int i = tid; i < N; i += 256) { float e = __expf(row[i] - mx); row[i] = e; sum += e; }
  sred[tid] = sum; __syncthreads();
  for (int k = 128; k > 0; k >>= 1) { if (tid < k) sred[tid] += sred[tid + k]; __syncthreads(); }
  float inv = 1.f / sred[0];
  for (int i = tid; i < N; i += 256) row[i] *= inv;
}

// ---------------------------------------------------------------------------
extern "C" void kernel_launch(void* const* d_in, const int* in_sizes, int n_in,
                              void* d_out, int out_size, void* d_ws, size_t ws_size,
                              hipStream_t stream) {
  const int*   tokens    = (const int*)d_in[0];
  const float* embed_w   = (const float*)d_in[1];
  const float* posn_w    = (const float*)d_in[2];
  const float* qkv_w     = (const float*)d_in[3];
  const float* out_w     = (const float*)d_in[4];
  const float* ln1_w     = (const float*)d_in[5];
  const float* ln1_b     = (const float*)d_in[6];
  const float* ln2_w     = (const float*)d_in[7];
  const float* ln2_b     = (const float*)d_in[8];
  const float* w1_w      = (const float*)d_in[9];
  const float* w1_b      = (const float*)d_in[10];
  const float* w2_w      = (const float*)d_in[11];
  const float* w2_b      = (const float*)d_in[12];
  const float* unembed_w = (const float*)d_in[13];
  const float* unembed_b = (const float*)d_in[14];
  float* out = (float*)d_out;

  // Workspace carve-out (~185 MB): bf16 weights + f32 activation buffers.
  char* base = (char*)d_ws;
  size_t off = 0;
  auto alloc = [&](size_t bytes) -> void* {
    void* p = base + off;
    off += (bytes + 255) & ~(size_t)255;
    return p;
  };
  __bf16* bq = (__bf16*)alloc((size_t)NL * HS * 3 * HS * sizeof(__bf16));
  __bf16* bo = (__bf16*)alloc((size_t)NL * DIM * DIM * sizeof(__bf16));
  __bf16* b1 = (__bf16*)alloc((size_t)NL * DIM * FF * sizeof(__bf16));
  __bf16* b2 = (__bf16*)alloc((size_t)NL * FF * DIM * sizeof(__bf16));
  __bf16* bu = (__bf16*)alloc((size_t)DIM * VOCP * sizeof(__bf16));  // padded
  float* x      = (float*)alloc((size_t)SEQ * DIM * sizeof(float));
  float* qkvbuf = (float*)alloc((size_t)SEQ * QKVSTR * sizeof(float));
  float* attnb  = (float*)alloc((size_t)SEQ * DIM * sizeof(float));
  float* tmp    = (float*)alloc((size_t)SEQ * DIM * sizeof(float));
  float* hbuf   = (float*)alloc((size_t)SEQ * FF * sizeof(float));

  // 1) Weights -> bf16 (one-time; unembed padded to VOCP columns).
  f32_to_bf16<<<1024, 256, 0, stream>>>(qkv_w, bq, NL * HS * 3 * HS);
  f32_to_bf16<<<1024, 256, 0, stream>>>(out_w, bo, NL * DIM * DIM);
  f32_to_bf16<<<2048, 256, 0, stream>>>(w1_w, b1, NL * DIM * FF);
  f32_to_bf16<<<2048, 256, 0, stream>>>(w2_w, b2, NL * FF * DIM);
  f32_to_bf16_pad<<<4096, 256, 0, stream>>>(unembed_w, bu, DIM, VOC, VOCP);

  // 2) Embedding + positional.
  embed_kernel<<<SEQ, 256, 0, stream>>>(tokens, embed_w, posn_w, x);

  // 3) Transformer layers.
  for (int l = 0; l < NL; ++l) {
    // Head-shared QKV: x viewed as [S*H, 64] (contiguous) @ [64, 192].
    wmma_gemm<0><<<dim3(2, (SEQ * NH) / 64), 256, 0, stream>>>(
        x, bq + (size_t)l * HS * 3 * HS, nullptr, qkvbuf,
        SEQ * NH, 3 * HS, HS, 3 * HS);
    // Causal flash attention.
    attn_kernel<<<dim3(NH, SEQ / 16), 32, 0, stream>>>(qkvbuf, attnb);
    // Output projection.
    wmma_gemm<0><<<dim3(DIM / 128, SEQ / 64), 256, 0, stream>>>(
        attnb, bo + (size_t)l * DIM * DIM, nullptr, tmp, SEQ, DIM, DIM, DIM);
    // x += LN(attn_proj)
    ln_residual<<<SEQ, 256, 0, stream>>>(tmp, ln1_w + l * DIM, ln1_b + l * DIM, x);
    // h = gelu(x @ W1 + b1)
    wmma_gemm<2><<<dim3(FF / 128, SEQ / 64), 256, 0, stream>>>(
        x, b1 + (size_t)l * DIM * FF, w1_b + l * FF, hbuf, SEQ, FF, DIM, FF);
    // t = h @ W2 + b2
    wmma_gemm<1><<<dim3(DIM / 128, SEQ / 64), 256, 0, stream>>>(
        hbuf, b2 + (size_t)l * FF * DIM, w2_b + l * DIM, tmp, SEQ, DIM, FF, DIM);
    // x += LN(t)
    ln_residual<<<SEQ, 256, 0, stream>>>(tmp, ln2_w + l * DIM, ln2_b + l * DIM, x);
  }

  // 4) Logits straight into d_out (B padded to VOCP), then in-place softmax.
  wmma_gemm<1><<<dim3(VOCP / 128, SEQ / 64), 256, 0, stream>>>(
      x, bu, unembed_b, out, SEQ, VOC, DIM, VOCP);
  softmax_rows<<<SEQ, 256, 0, stream>>>(out, VOC);
}
